// MobileMQA3D_21809843929709
// MI455X (gfx1250) — compile-verified, matrix operand order checked by reading, not run on hardware
//
#include <hip/hip_runtime.h>

// ---------------------------------------------------------------------------
// MobileMQA3D for MI455X (gfx1250, wave32): flash attention on v_wmma f16,
// double-buffered async-to-LDS staging (ASYNCcnt) for K and pre-transposed V.
// B=2, C=512, Ck=128, N=8192
// ---------------------------------------------------------------------------

typedef _Float16 v16h __attribute__((ext_vector_type(16)));
typedef _Float16 v8h  __attribute__((ext_vector_type(8)));
typedef float    v8f  __attribute__((ext_vector_type(8)));

#define CK   128
#define CIN  512
#define NSEQ 8192

union V16U { v16h v; v8h h[2]; };

// A/B fragment for v_wmma_f32_16x16x32_f16 from a row-major [row][K] tile:
// lane holds K = kb..kb+7 (elems 0..7) and kb+16..kb+23 (elems 8..15),
// kb = 0 (lanes 0-15) or 8 (lanes 16-31). base must point at [row][k0+kb].
__device__ __forceinline__ v16h load_frag(const _Float16* base) {
  V16U u;
  u.h[0] = *(const v8h*)(base);
  u.h[1] = *(const v8h*)(base + 16);
  return u.v;
}

__device__ __forceinline__ v8f wmma16(v16h a, v16h b, v8f c) {
  return __builtin_amdgcn_wmma_f32_16x16x32_f16(false, a, false, b,
                                                (short)0, c, false, false);
}

__device__ __forceinline__ v8f splat8(float x) {
  v8f v = {x, x, x, x, x, x, x, x};
  return v;
}

// 16B async DMA: global -> LDS, tracked by ASYNCcnt (CDNA5 path).
// Low 32 bits of a flat shared pointer are the wave-relative LDS offset
// (LDS aperture lives entirely in addr[63:32] per the ISA aperture check).
__device__ __forceinline__ void async_copy_b128(void* lds_dst,
                                                const void* gsrc) {
  unsigned lds = (unsigned)(unsigned long long)lds_dst;
  asm volatile("global_load_async_to_lds_b128 %0, %1, off"
               :: "v"(lds), "v"((unsigned long long)gsrc)
               : "memory");
}

__device__ __forceinline__ void wait_async0() {
  asm volatile("s_wait_asynccnt 0x0" ::: "memory");
}

// ---------------------------------------------------------------------------
// K1: weight prep. Wq/Wk/Wv -> f16 ([128][512] row-major, B-fragment native).
// Wo folded over the tile(1,1,4): Wo_eff[o][ck] = sum_j Wo[o][ck+128j], -> f16.
// ---------------------------------------------------------------------------
__global__ void prep_kernel(const float* __restrict__ Wq,
                            const float* __restrict__ Wk,
                            const float* __restrict__ Wv,
                            const float* __restrict__ Wo,
                            _Float16* __restrict__ Wqh,
                            _Float16* __restrict__ Wkh,
                            _Float16* __restrict__ Wvh,
                            _Float16* __restrict__ Woe) {
  int i = blockIdx.x * 256 + threadIdx.x;         // < 65536
  Wqh[i] = (_Float16)Wq[i];
  Wkh[i] = (_Float16)Wk[i];
  Wvh[i] = (_Float16)Wv[i];
  int o = i >> 7, ck = i & 127;                   // Wo_eff is [512][128]
  const float* r = Wo + o * CIN + ck;
  Woe[i] = (_Float16)(r[0] + r[128] + r[256] + r[384]);
}

// ---------------------------------------------------------------------------
// K2: projections. block = (ntile, b, p in {q,k,v}); 256 thr = 8 waves.
// Out[b][n][o] = sum_c W[o][c] * x[b][c][n] + bias[o], stored f16 [B][N][128].
// ---------------------------------------------------------------------------
__global__ __launch_bounds__(256)
void proj_kernel(const float* __restrict__ x,
                 const _Float16* __restrict__ Wqh,
                 const _Float16* __restrict__ Wkh,
                 const _Float16* __restrict__ Wvh,
                 const float* __restrict__ bq,
                 const float* __restrict__ bk,
                 const float* __restrict__ bv,
                 _Float16* __restrict__ Qh,
                 _Float16* __restrict__ Kh,
                 _Float16* __restrict__ Vh) {
  __shared__ _Float16 Xs[128][72];                // [n][c-chunk 64]

  const int tid  = threadIdx.x;
  const int lane = tid & 31, wave = tid >> 5;
  const int ntile = blockIdx.x, b = blockIdx.y, p = blockIdx.z;

  const _Float16* W    = (p == 0) ? Wqh : (p == 1) ? Wkh : Wvh;
  const float*    bias = (p == 0) ? bq  : (p == 1) ? bk  : bv;
  _Float16*       Out  = (p == 0) ? Qh  : (p == 1) ? Kh  : Vh;

  const int n0  = ntile * 128;
  const int row = lane & 15;
  const int kb  = (lane < 16) ? 0 : 8;
  const int hi8 = kb;
  const float* xb = x + (size_t)b * CIN * NSEQ;

  v8f acc[8];
#pragma unroll
  for (int ot = 0; ot < 8; ++ot) acc[ot] = splat8(0.f);

  for (int kc = 0; kc < 8; ++kc) {                // K chunks of 64 over C=512
    const int c0 = kc * 64;
    __syncthreads();
    for (int i = tid; i < 64 * 128; i += 256) {   // coalesced along n
      int c = i >> 7, n = i & 127;
      Xs[n][c] = (_Float16)xb[(size_t)(c0 + c) * NSEQ + n0 + n];
    }
    __syncthreads();

    v16h a0 = load_frag(&Xs[wave * 16 + row][kb]);
    v16h a1 = load_frag(&Xs[wave * 16 + row][32 + kb]);
#pragma unroll
    for (int ot = 0; ot < 8; ++ot) {
      const _Float16* wr = W + (size_t)(ot * 16 + row) * CIN + c0;
      acc[ot] = wmma16(a0, load_frag(wr + kb), acc[ot]);
      acc[ot] = wmma16(a1, load_frag(wr + 32 + kb), acc[ot]);
    }
  }

#pragma unroll
  for (int ot = 0; ot < 8; ++ot) {
    int o = ot * 16 + row;
    float bv_ = bias[o];
#pragma unroll
    for (int r = 0; r < 8; ++r) {
      int n = n0 + wave * 16 + r + hi8;
      Out[((size_t)b * NSEQ + n) * CK + o] = (_Float16)(acc[ot][r] + bv_);
    }
  }
}

// ---------------------------------------------------------------------------
// K2.5: one-time V transpose: Vh [B][N][128] -> Vt [B][128][N].
// Done once so attention staging is pure async DMA in B-native layout
// (instead of 64 q-tile workgroups each re-transposing V 128 times).
// ---------------------------------------------------------------------------
__global__ __launch_bounds__(256)
void transposeV_kernel(const _Float16* __restrict__ Vh,
                       _Float16* __restrict__ Vt) {
  __shared__ _Float16 Ls[64][136];
  const int tid = threadIdx.x;
  const int b = blockIdx.y, n0 = blockIdx.x * 64;
  for (int i = tid; i < 64 * 128; i += 256) {     // coalesced read
    int n = i >> 7, o = i & 127;
    Ls[n][o] = Vh[((size_t)b * NSEQ + n0 + n) * CK + o];
  }
  __syncthreads();
  for (int i = tid; i < 128 * 64; i += 256) {     // coalesced write along n
    int o = i >> 6, nn = i & 63;
    Vt[((size_t)b * CK + o) * NSEQ + n0 + nn] = Ls[nn][o];
  }
}

// ---------------------------------------------------------------------------
// K3: flash attention. block = (qtile of 128 rows, b); 8 waves x 16 rows.
// Q fragments persistent in VGPRs; K and V(pre-transposed) staged to LDS by
// double-buffered global_load_async_to_lds_b128 (ASYNCcnt), overlapping the
// next chunk's DMA with this chunk's 32 WMMA per wave. One barrier per chunk.
// ---------------------------------------------------------------------------
__global__ __launch_bounds__(256)
void attn_kernel(const _Float16* __restrict__ Qh,
                 const _Float16* __restrict__ Kh,
                 const _Float16* __restrict__ Vt,
                 _Float16* __restrict__ Oh) {
  __shared__ _Float16 Ks[2][64][136];             // [buf][key][c]   34 KB
  __shared__ _Float16 Vs[2][128][72];             // [buf][o][key]   36 KB
  __shared__ _Float16 Pb[8][16][72];              // per-wave P      18 KB

  const int tid  = threadIdx.x;
  const int lane = tid & 31, wave = tid >> 5;
  const int b = blockIdx.y, n0 = blockIdx.x * 128;
  const int row = lane & 15;
  const int kb  = (lane < 16) ? 0 : 8;
  const int hi8 = kb;

  const _Float16* Qb  = Qh + (size_t)b * NSEQ * CK;
  const _Float16* Kb  = Kh + (size_t)b * NSEQ * CK;
  const _Float16* Vtb = Vt + (size_t)b * CK * NSEQ;

  // persistent Q A-fragments for this wave's 16 query rows (K = 128)
  v16h qf[4];
  const _Float16* qr = Qb + (size_t)(n0 + wave * 16 + row) * CK;
#pragma unroll
  for (int f = 0; f < 4; ++f) qf[f] = load_frag(qr + 32 * f + kb);

  v8f acc[8];
#pragma unroll
  for (int ot = 0; ot < 8; ++ot) acc[ot] = splat8(0.f);
  v8f mstate = splat8(-1e30f);
  v8f lstate = splat8(0.f);
  const float c2 = 0.08838834764831845f * 1.4426950408889634f; // rsqrt(128)*log2e

  // ---- prologue: DMA chunk 0 into buffer 0 ----
  for (int i = tid; i < 1024; i += 256) {         // K: 64 rows x 16 x 16B
    int kr = i >> 4, cg = i & 15;
    async_copy_b128(&Ks[0][kr][cg * 8], Kb + (size_t)kr * CK + cg * 8);
  }
  for (int i = tid; i < 1024; i += 256) {         // V: 128 rows x 8 x 16B
    int o = i >> 3, cg = i & 7;
    async_copy_b128(&Vs[0][o][cg * 8], Vtb + (size_t)o * NSEQ + cg * 8);
  }
  wait_async0();
  __syncthreads();

  for (int kc = 0; kc < NSEQ / 64; ++kc) {
    const int cur = kc & 1;

    // ---- issue DMA for chunk kc+1 into the other buffer (overlaps compute)
    if (kc + 1 < NSEQ / 64) {
      const int k1 = (kc + 1) * 64, nb = cur ^ 1;
      for (int i = tid; i < 1024; i += 256) {
        int kr = i >> 4, cg = i & 15;
        async_copy_b128(&Ks[nb][kr][cg * 8],
                        Kb + (size_t)(k1 + kr) * CK + cg * 8);
      }
      for (int i = tid; i < 1024; i += 256) {
        int o = i >> 3, cg = i & 7;
        async_copy_b128(&Vs[nb][o][cg * 8],
                        Vtb + (size_t)o * NSEQ + k1 + cg * 8);
      }
    }

    // ---- S = Q K^T : 4 key tiles of 16, K-dim = 128 ----
    v8f s[4];
#pragma unroll
    for (int kt = 0; kt < 4; ++kt) {
      v8f sa = splat8(0.f);
#pragma unroll
      for (int f = 0; f < 4; ++f)
        sa = wmma16(qf[f], load_frag(&Ks[cur][kt * 16 + row][32 * f + kb]), sa);
      s[kt] = sa;
    }

    // ---- online softmax (rows live in 16-lane halves under wave32) ----
    v8f tmax = __builtin_elementwise_max(__builtin_elementwise_max(s[0], s[1]),
                                         __builtin_elementwise_max(s[2], s[3]));
#pragma unroll
    for (int m = 1; m <= 8; m <<= 1)
#pragma unroll
      for (int r = 0; r < 8; ++r)
        tmax[r] = fmaxf(tmax[r], __shfl_xor(tmax[r], m, 32));
    v8f mnew = __builtin_elementwise_max(mstate, tmax);
    v8f alpha;
#pragma unroll
    for (int r = 0; r < 8; ++r) alpha[r] = exp2f((mstate[r] - mnew[r]) * c2);
#pragma unroll
    for (int ot = 0; ot < 8; ++ot) acc[ot] *= alpha;
    v8f psum = splat8(0.f);
#pragma unroll
    for (int kt = 0; kt < 4; ++kt) {
#pragma unroll
      for (int r = 0; r < 8; ++r) s[kt][r] = exp2f((s[kt][r] - mnew[r]) * c2);
      psum += s[kt];
    }
#pragma unroll
    for (int m = 1; m <= 8; m <<= 1)
#pragma unroll
      for (int r = 0; r < 8; ++r) psum[r] += __shfl_xor(psum[r], m, 32);
    lstate = lstate * alpha + psum;
    mstate = mnew;

    // ---- P: C/D layout -> A layout via wave-private LDS (in-order per wave)
    _Float16(*Pw)[72] = Pb[wave];
#pragma unroll
    for (int kt = 0; kt < 4; ++kt)
#pragma unroll
      for (int r = 0; r < 8; ++r)
        Pw[r + hi8][kt * 16 + row] = (_Float16)s[kt][r];

    v16h pa0 = load_frag(&Pw[row][kb]);
    v16h pa1 = load_frag(&Pw[row][32 + kb]);
#pragma unroll
    for (int ot = 0; ot < 8; ++ot) {
      acc[ot] = wmma16(pa0, load_frag(&Vs[cur][ot * 16 + row][kb]), acc[ot]);
      acc[ot] = wmma16(pa1, load_frag(&Vs[cur][ot * 16 + row][32 + kb]), acc[ot]);
    }

    wait_async0();                                // next chunk landed
    __syncthreads();                              // all waves done with cur
  }

  v8f linv;
#pragma unroll
  for (int r = 0; r < 8; ++r) linv[r] = 1.0f / lstate[r];
#pragma unroll
  for (int ot = 0; ot < 8; ++ot)
#pragma unroll
    for (int r = 0; r < 8; ++r) {
      int n = n0 + wave * 16 + r + hi8;
      Oh[((size_t)b * NSEQ + n) * CK + ot * 16 + row] =
          (_Float16)(acc[ot][r] * linv[r]);
    }
}

// ---------------------------------------------------------------------------
// K4: output projection + bias + residual.
// y[b][o][n] = sum_ck Wo_eff[o][ck]*O[b][n][ck] + bo[o] + x[b][o][n]
// ---------------------------------------------------------------------------
__global__ __launch_bounds__(256)
void outproj_kernel(const _Float16* __restrict__ Oh,
                    const _Float16* __restrict__ Woe,
                    const float* __restrict__ bo,
                    const float* __restrict__ x,
                    float* __restrict__ y) {
  const int tid  = threadIdx.x;
  const int lane = tid & 31, wave = tid >> 5;
  const int b = blockIdx.y, n0 = blockIdx.x * 128;
  const int row = lane & 15;
  const int kb  = (lane < 16) ? 0 : 8;
  const int hi8 = kb;

  const _Float16* orow = Oh + ((size_t)b * NSEQ + n0 + wave * 16 + row) * CK;
  v16h of[4];
#pragma unroll
  for (int f = 0; f < 4; ++f) of[f] = load_frag(orow + 32 * f + kb);

  for (int ot = 0; ot < 32; ++ot) {               // 512 output channels
    v8f acc = splat8(0.f);
    const _Float16* wr = Woe + (size_t)(ot * 16 + row) * CK;
#pragma unroll
    for (int f = 0; f < 4; ++f)
      acc = wmma16(load_frag(wr + 32 * f + kb), of[f], acc);
    int n = n0 + wave * 16 + row;
#pragma unroll
    for (int r = 0; r < 8; ++r) {
      int o = ot * 16 + r + hi8;
      size_t idx = ((size_t)b * CIN + o) * NSEQ + n;
      y[idx] = acc[r] + bo[o] + x[idx];
    }
  }
}

// ---------------------------------------------------------------------------
extern "C" void kernel_launch(void* const* d_in, const int* in_sizes, int n_in,
                              void* d_out, int out_size, void* d_ws,
                              size_t ws_size, hipStream_t stream) {
  const float* x  = (const float*)d_in[0];
  const float* Wq = (const float*)d_in[1];
  const float* bq = (const float*)d_in[2];
  const float* Wk = (const float*)d_in[3];
  const float* bk = (const float*)d_in[4];
  const float* Wv = (const float*)d_in[5];
  const float* bv = (const float*)d_in[6];
  const float* Wo = (const float*)d_in[7];
  const float* bo = (const float*)d_in[8];
  float* out = (float*)d_out;

  const size_t NTOK = (size_t)2 * NSEQ * CK;
  _Float16* Qh  = (_Float16*)d_ws;
  _Float16* Kh  = Qh + NTOK;
  _Float16* Vh  = Kh + NTOK;
  _Float16* Ohp = Vh + NTOK;
  _Float16* Wqh = Ohp + NTOK;
  _Float16* Wkh = Wqh + 128 * 512;
  _Float16* Wvh = Wkh + 128 * 512;
  _Float16* Woe = Wvh + 128 * 512;
  _Float16* Vtp = Woe + 128 * 512;                // [B][128][N], 4 MB

  prep_kernel<<<256, 256, 0, stream>>>(Wq, Wk, Wv, Wo, Wqh, Wkh, Wvh, Woe);
  proj_kernel<<<dim3(64, 2, 3), 256, 0, stream>>>(x, Wqh, Wkh, Wvh, bq, bk, bv,
                                                  Qh, Kh, Vh);
  transposeV_kernel<<<dim3(128, 2), 256, 0, stream>>>(Vh, Vtp);
  attn_kernel<<<dim3(64, 2), 256, 0, stream>>>(Qh, Kh, Vtp, Ohp);
  outproj_kernel<<<dim3(64, 2), 256, 0, stream>>>(Ohp, Woe, bo, x, out);
}
